// Decoder_81621558493622
// MI455X (gfx1250) — compile-verified, matrix operand order checked by reading, not run on hardware
//
#include <hip/hip_runtime.h>
#include <hip/hip_bf16.h>

// ---------------------------------------------------------------------------
// Attention-LSTM decoder for MI455X (gfx1250, wave32, WMMA).
// Heavy GEMMs run on v_wmma_f32_16x16x32_f16 with 2x2 register tile blocking
// (one wave owns a 32x32 output patch -> 4 independent accumulators, halved
// L2 traffic on the weight stream) + global_prefetch on the B stream.
// ---------------------------------------------------------------------------

#define Bc   32
#define Tc   64
#define Sc   128
#define Vc   32000
#define EMBc 512
#define Hc   1024
#define Ec   1024

typedef __attribute__((ext_vector_type(16))) _Float16 v16h;
typedef __attribute__((ext_vector_type(8)))  float    v8f;

// ---------------------------------------------------------------------------
// f16 GEMM:  C[m,n] = sum_k A[m,k] * W[n,k]  (+ bias[n])
// A: M x K row-major (lda), W: N x K row-major (ldb), C: row stride ldc.
// Block = 128 threads = 4 waves. Each wave computes a 32x32 output patch
// (2 M-tiles x 2 N-tiles, 4 accumulators) -> each A/B fragment feeds 2 WMMAs.
// Requirements: M % 32 == 0, N % 128 == 0, K % 128 == 0,
// lda/ldb multiples of 8 halves (16B-aligned uint4 fragment loads).
// ---------------------------------------------------------------------------
__global__ void k_gemm_f16(const _Float16* __restrict__ A,
                           const _Float16* __restrict__ W,
                           float* __restrict__ C,
                           const float* __restrict__ bias,
                           int M, int N, int K,
                           int lda, int ldb, long ldc) {
    const int lane = threadIdx.x & 31;
    const int wave = threadIdx.x >> 5;
    const int nBase = (blockIdx.x * 4 + wave) * 32;   // two 16-col tiles
    const int mBase = blockIdx.y * 32;                // two 16-row tiles
    if (nBase >= N || mBase >= M) return;

    const int half = lane >> 4;          // 0: lanes 0-15, 1: lanes 16-31
    const int l16  = lane & 15;

    const _Float16* A0 = A + (long)(mBase      + l16) * lda;
    const _Float16* A1 = A + (long)(mBase + 16 + l16) * lda;
    const _Float16* W0 = W + (long)(nBase      + l16) * ldb;
    const _Float16* W1 = W + (long)(nBase + 16 + l16) * ldb;

    v8f acc00 = {}, acc01 = {}, acc10 = {}, acc11 = {};

    for (int kb0 = 0; kb0 < K; kb0 += 128) {
        // Prefetch next super-chunk of the weight stream (speculative;
        // harmless past the end) -> global_prefetch path.
        __builtin_prefetch(W0 + kb0 + 128, 0, 1);
        __builtin_prefetch(W1 + kb0 + 128, 0, 1);
        #pragma unroll
        for (int kk = 0; kk < 128; kk += 32) {
            const int kb = kb0 + kk;
            union { uint4 q[2]; v16h v; } a0, a1, b0, b1;
            const uint4* ap0 = reinterpret_cast<const uint4*>(A0 + kb);
            const uint4* ap1 = reinterpret_cast<const uint4*>(A1 + kb);
            const uint4* bp0 = reinterpret_cast<const uint4*>(W0 + kb);
            const uint4* bp1 = reinterpret_cast<const uint4*>(W1 + kb);
            // Lane fragment: K = half*8 + {0..7} and K = 16 + half*8 + {0..7}
            a0.q[0] = ap0[half]; a0.q[1] = ap0[2 + half];
            a1.q[0] = ap1[half]; a1.q[1] = ap1[2 + half];
            b0.q[0] = bp0[half]; b0.q[1] = bp0[2 + half];
            b1.q[0] = bp1[half]; b1.q[1] = bp1[2 + half];
            acc00 = __builtin_amdgcn_wmma_f32_16x16x32_f16(
                false, a0.v, false, b0.v, (short)0, acc00, false, false);
            acc01 = __builtin_amdgcn_wmma_f32_16x16x32_f16(
                false, a0.v, false, b1.v, (short)0, acc01, false, false);
            acc10 = __builtin_amdgcn_wmma_f32_16x16x32_f16(
                false, a1.v, false, b0.v, (short)0, acc10, false, false);
            acc11 = __builtin_amdgcn_wmma_f32_16x16x32_f16(
                false, a1.v, false, b1.v, (short)0, acc11, false, false);
        }
    }

    const int c0 = nBase + l16, c1 = nBase + 16 + l16;
    const float bv0 = bias ? bias[c0] : 0.0f;
    const float bv1 = bias ? bias[c1] : 0.0f;
    const int r0 = mBase + 8 * half;        // C/D layout: M = r + 8*half
    const int r1 = mBase + 16 + 8 * half;
    #pragma unroll
    for (int r = 0; r < 8; ++r) {
        C[(long)(r0 + r) * ldc + c0] = acc00[r] + bv0;
        C[(long)(r0 + r) * ldc + c1] = acc01[r] + bv1;
        C[(long)(r1 + r) * ldc + c0] = acc10[r] + bv0;
        C[(long)(r1 + r) * ldc + c1] = acc11[r] + bv1;
    }
}

// --------------------------- helper kernels --------------------------------
__global__ void k_cvt_f16(const float* __restrict__ s, _Float16* __restrict__ d, long n) {
    for (long i = blockIdx.x * (long)blockDim.x + threadIdx.x; i < n;
         i += (long)gridDim.x * blockDim.x)
        d[i] = (_Float16)s[i];
}

// Wcat[n, 0:1536] = W_ih[n,:],  Wcat[n, 1536:2560] = W_hh[n,:]   (f16)
__global__ void k_wcat(const float* __restrict__ wih, const float* __restrict__ whh,
                       _Float16* __restrict__ d) {
    const long n = (long)4096 * 2560;
    for (long i = blockIdx.x * (long)blockDim.x + threadIdx.x; i < n;
         i += (long)gridDim.x * blockDim.x) {
        int row = (int)(i / 2560), k = (int)(i % 2560);
        float v = (k < 1536) ? wih[(long)row * 1536 + k]
                             : whh[(long)row * 1024 + (k - 1536)];
        d[i] = (_Float16)v;
    }
}

// embedded (f16): emb16[bt, e] = emb[target[bt], e]
__global__ void k_embed(const int* __restrict__ tgt, const float* __restrict__ emb,
                        _Float16* __restrict__ d) {
    const long n = (long)Bc * Tc * EMBc;
    for (long i = blockIdx.x * (long)blockDim.x + threadIdx.x; i < n;
         i += (long)gridDim.x * blockDim.x) {
        long bt = i / EMBc; int e = (int)(i % EMBc);
        d[i] = (_Float16)emb[(long)tgt[bt] * EMBc + e];
    }
}

__global__ void k_bias4(const float* __restrict__ bih, const float* __restrict__ bhh,
                        float* __restrict__ d) {
    int i = blockIdx.x * blockDim.x + threadIdx.x;
    if (i < 4 * Hc) d[i] = bih[i] + bhh[i];
}

__global__ void k_init(_Float16* __restrict__ h16, float* __restrict__ c) {
    int i = blockIdx.x * blockDim.x + threadIdx.x;
    if (i < Bc * Hc) { h16[i] = (_Float16)0.0f; c[i] = 0.0f; }
}

__device__ __forceinline__ float sigf(float x) { return 1.0f / (1.0f + expf(-x)); }

// Attention + assemble lstm_in = [x_t | context | h] (f16). One block per batch.
__global__ void k_attn(const float* __restrict__ hWh,       // (B,H)
                       const float* __restrict__ enc_proj,  // (B,S,H)
                       const float* __restrict__ enc,       // (B,S,E) fp32 input
                       const float* __restrict__ vvec,      // (H)
                       const _Float16* __restrict__ emb16,  // (B,T,EMB)
                       const _Float16* __restrict__ h16,    // (B,H)
                       _Float16* __restrict__ lstm_in16,    // (B, EMB+E+H = 2560)
                       int t) {
    __shared__ float s_sc[Sc];
    __shared__ float s_red[256];
    const int b = blockIdx.x, tid = threadIdx.x;
    const int lane = tid & 31, wv = tid >> 5;       // 8 waves of 32 (wave32)

    // scores[s] = v . tanh(hWh[b,:] + enc_proj[b,s,:])
    for (int s = wv; s < Sc; s += 8) {
        const float* ep = enc_proj + ((long)b * Sc + s) * Hc;
        float p = 0.0f;
        for (int h = lane; h < Hc; h += 32)
            p += vvec[h] * tanhf(hWh[b * Hc + h] + ep[h]);
        for (int off = 16; off > 0; off >>= 1) p += __shfl_down(p, off, 32);
        if (lane == 0) s_sc[s] = p;
    }
    __syncthreads();

    // softmax over S
    float lm = -1e30f;
    for (int s = tid; s < Sc; s += 256) lm = fmaxf(lm, s_sc[s]);
    s_red[tid] = lm; __syncthreads();
    for (int st = 128; st > 0; st >>= 1) {
        if (tid < st) s_red[tid] = fmaxf(s_red[tid], s_red[tid + st]);
        __syncthreads();
    }
    const float mx = s_red[0]; __syncthreads();
    float ls = 0.0f;
    for (int s = tid; s < Sc; s += 256) { float e = expf(s_sc[s] - mx); s_sc[s] = e; ls += e; }
    s_red[tid] = ls; __syncthreads();
    for (int st = 128; st > 0; st >>= 1) {
        if (tid < st) s_red[tid] += s_red[tid + st];
        __syncthreads();
    }
    const float inv = 1.0f / s_red[0]; __syncthreads();

    // context[e] = sum_s w[s] * enc[b,s,e]  -> lstm_in[512 + e]
    _Float16* dst = lstm_in16 + (long)b * 2560;
    for (int e = tid; e < Ec; e += 256) {
        const float* eb = enc + ((long)b * Sc) * Ec + e;
        float a = 0.0f;
        for (int s = 0; s < Sc; ++s) a += s_sc[s] * eb[(long)s * Ec];
        dst[EMBc + e] = (_Float16)(a * inv);
    }
    // x_t part
    for (int e = tid; e < EMBc; e += 256)
        dst[e] = emb16[((long)b * Tc + t) * EMBc + e];
    // h part (for fused W_hh GEMM)
    for (int j = tid; j < Hc; j += 256)
        dst[EMBc + Ec + j] = h16[b * Hc + j];
}

// LSTM pointwise: gates (B,4H) -> c (f32), h16 (f16)
__global__ void k_lstm(const float* __restrict__ gates,
                       float* __restrict__ c, _Float16* __restrict__ h16) {
    int i = blockIdx.x * blockDim.x + threadIdx.x;
    if (i >= Bc * Hc) return;
    int b = i >> 10, n = i & (Hc - 1);
    const float* g = gates + (long)b * 4 * Hc;
    float gi = g[n], gf = g[Hc + n], gg = g[2 * Hc + n], go = g[3 * Hc + n];
    float cn = sigf(gf) * c[i] + sigf(gi) * tanhf(gg);
    float hn = sigf(go) * tanhf(cn);
    c[i] = cn;
    h16[i] = (_Float16)hn;
}

// ---------------------------------------------------------------------------
extern "C" void kernel_launch(void* const* d_in, const int* in_sizes, int n_in,
                              void* d_out, int out_size, void* d_ws, size_t ws_size,
                              hipStream_t stream) {
    const int*   target = (const int*)  d_in[0];
    const float* enc    = (const float*)d_in[1];   // (B,S,E)
    const float* emb    = (const float*)d_in[2];   // (V,EMB)
    const float* attnW  = (const float*)d_in[3];   // (H, H+E)
    const float* attnb  = (const float*)d_in[4];   // (H)
    const float* vvec   = (const float*)d_in[5];   // (H)
    const float* W_ih   = (const float*)d_in[6];   // (4H, EMB+E)
    const float* W_hh   = (const float*)d_in[7];   // (4H, H)
    const float* b_ih   = (const float*)d_in[8];
    const float* b_hh   = (const float*)d_in[9];
    const float* fc_W   = (const float*)d_in[10];  // (V, H)
    const float* fc_b   = (const float*)d_in[11];
    float* out = (float*)d_out;                    // (B, T, V)

    // ---- workspace carve (256B aligned) ----
    char* p = (char*)d_ws;
    auto carve = [&](size_t bytes) {
        char* r = p;
        p += (bytes + 255) & ~(size_t)255;
        return (void*)r;
    };
    _Float16* attnW16 = (_Float16*)carve((size_t)Hc * (Hc + Ec) * 2);   //  4.2 MB
    _Float16* Wcat16  = (_Float16*)carve((size_t)4 * Hc * 2560 * 2);    // 21.0 MB
    _Float16* fcW16   = (_Float16*)carve((size_t)Vc * Hc * 2);          // 65.5 MB
    _Float16* enc16   = (_Float16*)carve((size_t)Bc * Sc * Ec * 2);     //  8.4 MB
    _Float16* emb16   = (_Float16*)carve((size_t)Bc * Tc * EMBc * 2);   //  2.1 MB
    float*    encproj = (float*)   carve((size_t)Bc * Sc * Hc * 4);     // 16.8 MB
    float*    bias4   = (float*)   carve((size_t)4 * Hc * 4);
    _Float16* h16     = (_Float16*)carve((size_t)Bc * Hc * 2);
    float*    cbuf    = (float*)   carve((size_t)Bc * Hc * 4);
    float*    hWh     = (float*)   carve((size_t)Bc * Hc * 4);
    _Float16* lin16   = (_Float16*)carve((size_t)Bc * 2560 * 2);
    float*    gates   = (float*)   carve((size_t)Bc * 4 * Hc * 4);
    (void)ws_size; (void)in_sizes; (void)n_in; (void)out_size;

    // ---- one-time (per call) conversions ----
    k_cvt_f16<<<2048, 256, 0, stream>>>(attnW, attnW16, (long)Hc * (Hc + Ec));
    k_cvt_f16<<<2048, 256, 0, stream>>>(enc,   enc16,   (long)Bc * Sc * Ec);
    k_cvt_f16<<<2048, 256, 0, stream>>>(fc_W,  fcW16,   (long)Vc * Hc);
    k_wcat  <<<2048, 256, 0, stream>>>(W_ih, W_hh, Wcat16);
    k_embed <<<1024, 256, 0, stream>>>(target, emb, emb16);
    k_bias4 <<<16,   256, 0, stream>>>(b_ih, b_hh, bias4);
    k_init  <<<128,  256, 0, stream>>>(h16, cbuf);

    // enc_proj = enc @ We^T + attn_b : M=B*S=4096, N=H, K=E
    {
        dim3 grid(Hc / 128, (Bc * Sc) / 32);
        k_gemm_f16<<<grid, 128, 0, stream>>>(enc16, attnW16 + Hc /*We cols*/,
                                             encproj, attnb,
                                             Bc * Sc, Hc, Ec,
                                             Ec, Hc + Ec, (long)Hc);
    }

    // ---- sequential decode ----
    for (int t = 0; t < Tc; ++t) {
        // hWh = h @ Wh^T : M=32, N=H, K=H
        {
            dim3 grid(Hc / 128, Bc / 32);
            k_gemm_f16<<<grid, 128, 0, stream>>>(h16, attnW16, hWh, nullptr,
                                                 Bc, Hc, Hc,
                                                 Hc, Hc + Ec, (long)Hc);
        }
        // attention + build lstm_in = [x_t | context | h]
        k_attn<<<Bc, 256, 0, stream>>>(hWh, encproj, enc, vvec, emb16, h16, lin16, t);

        // gates = lstm_in @ [W_ih|W_hh]^T + (b_ih+b_hh) : M=32, N=4H, K=2560
        {
            dim3 grid((4 * Hc) / 128, Bc / 32);
            k_gemm_f16<<<grid, 128, 0, stream>>>(lin16, Wcat16, gates, bias4,
                                                 Bc, 4 * Hc, 2560,
                                                 2560, 2560, (long)(4 * Hc));
        }
        // LSTM pointwise -> h16, c
        k_lstm<<<(Bc * Hc + 255) / 256, 256, 0, stream>>>(gates, cbuf, h16);

        // logits = h @ fc_W^T + fc_b -> out[:, t, :] : M=32, N=V, K=H
        {
            dim3 grid(Vc / 128, Bc / 32);
            k_gemm_f16<<<grid, 128, 0, stream>>>(h16, fcW16,
                                                 out + (long)t * Vc, fc_b,
                                                 Bc, Vc, Hc,
                                                 Hc, Hc, (long)Tc * Vc);
        }
    }
}
